// LlamaTernaryMLP_63883343560961
// MI455X (gfx1250) — compile-verified, matrix operand order checked by reading, not run on hardware
//
#include <hip/hip_runtime.h>
#include <hip/hip_bf16.h>
#include <math.h>

// ---------------------------------------------------------------------------
// LlamaTernaryMLP for MI455X (gfx1250, wave32, WMMA).
//   gate = x @ Wg^T * ag ; up = x @ Wu^T * au ; h = silu(gate)*up
//   out  = h @ Wd^T * ad
// bf16 WMMA (v_wmma_f32_16x16x32_bf16), f32 accumulate; ternary weights are
// exact in bf16. Software-pipelined: double-buffered LDS, global loads for
// stage s+1 in flight during WMMA of stage s; kernel 2 streams its bf16
// A-tile with CDNA5 async global->LDS (ASYNCcnt) overlapped with compute.
// ---------------------------------------------------------------------------

typedef __bf16 v16bf __attribute__((ext_vector_type(16)));
typedef __bf16 v8bf  __attribute__((ext_vector_type(8)));
typedef __bf16 bf4   __attribute__((ext_vector_type(4)));
typedef float  v8f   __attribute__((ext_vector_type(8)));

#define HIDDEN 4096
#define INTER  11008
#define BM 128          // C-tile rows per workgroup
#define BN 128          // C-tile cols per workgroup
#define BK 32           // K slab per LDS stage (one K=32 WMMA step)
#define STR (BK + 8)    // padded LDS row stride (bf16), 80 B: 16B-aligned

// 16x32 bf16 WMMA operand fragment from an LDS tile.
// Wave32 layout (ISA 7.12.2): lanes 0-15 = rows, K {0..7,16..23};
// lanes 16-31 = rows, K {8..15,24..31}. Two 16-byte ds loads.
__device__ __forceinline__ v16bf frag_ld(const __bf16* tile, int lane) {
  const int r  = lane & 15;
  const int ko = (lane & 16) ? 8 : 0;
  v16bf f;
  ((v8bf*)&f)[0] = *(const v8bf*)(tile + r * STR + ko);
  ((v8bf*)&f)[1] = *(const v8bf*)(tile + r * STR + ko + 16);
  return f;
}

__device__ __forceinline__ v8f wmma_bf16(v16bf a, v16bf b, v8f c) {
  return __builtin_amdgcn_wmma_f32_16x16x32_bf16(
      /*neg_a=*/false, a, /*neg_b=*/false, b,
      /*c_mod=*/(short)0, c, /*reuse_a=*/false, /*reuse_b=*/false);
}

// CDNA5 async copy: 16 B global -> LDS, tracked on ASYNCcnt (no VGPR staging).
__device__ __forceinline__ void async_g2l_b128(const void* lds_ptr,
                                               const void* gptr) {
  unsigned           lds_off = (unsigned)(unsigned long long)lds_ptr; // LDS byte offset
  unsigned long long gaddr   = (unsigned long long)gptr;
  asm volatile("global_load_async_to_lds_b128 %0, %1, off"
               :: "v"(lds_off), "v"(gaddr) : "memory");
}
__device__ __forceinline__ void wait_async0() {
  asm volatile("s_wait_asynccnt 0x0" ::: "memory");
}

__device__ __forceinline__ bf4 cvt_bf4(float4 v) {
  bf4 p = { (__bf16)v.x, (__bf16)v.y, (__bf16)v.z, (__bf16)v.w };
  return p;
}

// ------------------------- Kernel 1: gate/up + SwiGLU -----------------------
__global__ __launch_bounds__(256) void gateup_swiglu_kernel(
    const float* __restrict__ x,  const float* __restrict__ Wg,
    const float* __restrict__ Wu, const float* __restrict__ ag,
    const float* __restrict__ au, __bf16* __restrict__ h) {
  __shared__ __bf16 As[2][BM * STR];   // x tile,  BM x BK, double-buffered
  __shared__ __bf16 Bg[2][BN * STR];   // Wg tile, BN x BK
  __shared__ __bf16 Bu[2][BN * STR];   // Wu tile, BN x BK

  const int tid   = threadIdx.x;
  const int lane  = tid & 31;
  const int wave  = tid >> 5;       // 8 waves: 4 (M) x 2 (N)
  const int wm    = wave >> 1;      // 0..3 : 32-row band
  const int wn    = wave & 1;       // 0..1 : 64-col band
  const int mBase = blockIdx.y * BM;
  const int nBase = blockIdx.x * BN;

  v8f accG[2][4] = {};
  v8f accU[2][4] = {};

  float4 rx[4], rg[4], ru[4];       // stage registers (in flight during WMMA)

  // BM*BK/4 = 1024 float4 / 256 thr = 4 each; 8 float4 per 32-wide row.
  auto ld_glb = [&](int k0) {
#pragma unroll
    for (int i = 0; i < 4; ++i) {
      int c = tid + i * 256;
      int row = c >> 3, c4 = c & 7;
      rx[i] = *(const float4*)(x + (size_t)(mBase + row) * HIDDEN + k0 + c4 * 4);
      size_t off = (size_t)(nBase + row) * HIDDEN + k0 + c4 * 4;
      rg[i] = *(const float4*)(Wg + off);
      ru[i] = *(const float4*)(Wu + off);
    }
  };
  auto st_lds = [&](int buf) {
#pragma unroll
    for (int i = 0; i < 4; ++i) {
      int c = tid + i * 256;
      int row = c >> 3, c4 = c & 7;
      *(bf4*)(As[buf] + row * STR + c4 * 4) = cvt_bf4(rx[i]);
      *(bf4*)(Bg[buf] + row * STR + c4 * 4) = cvt_bf4(rg[i]);
      *(bf4*)(Bu[buf] + row * STR + c4 * 4) = cvt_bf4(ru[i]);
    }
  };
  auto compute = [&](int buf) {
    v16bf a0 = frag_ld(As[buf] + (wm * 32 + 0)  * STR, lane);
    v16bf a1 = frag_ld(As[buf] + (wm * 32 + 16) * STR, lane);
#pragma unroll
    for (int j = 0; j < 4; ++j) {
      v16bf bg = frag_ld(Bg[buf] + (wn * 64 + j * 16) * STR, lane);
      v16bf bu = frag_ld(Bu[buf] + (wn * 64 + j * 16) * STR, lane);
      accG[0][j] = wmma_bf16(a0, bg, accG[0][j]);
      accG[1][j] = wmma_bf16(a1, bg, accG[1][j]);
      accU[0][j] = wmma_bf16(a0, bu, accU[0][j]);
      accU[1][j] = wmma_bf16(a1, bu, accU[1][j]);
    }
  };

  // ---- software pipeline over K (128 stages, unrolled x2 for static bufs)
  ld_glb(0);
  st_lds(0);
  __syncthreads();
  const int nStages = HIDDEN / BK;              // 128 (even)
  for (int s = 0; s < nStages; s += 2) {
    ld_glb((s + 1) * BK);                       // next stage in flight
    compute(0);                                 // WMMA on buf 0
    st_lds(1);                                  // land stage s+1 in buf 1
    __syncthreads();
    if (s + 2 < nStages) ld_glb((s + 2) * BK);
    compute(1);                                 // WMMA on buf 1
    if (s + 2 < nStages) st_lds(0);
    __syncthreads();
  }

  // --- epilogue: scales + SiLU*up, write h as bf16.
  // C/D layout: VGPR v -> M=v (lanes 0-15) / M=8+v (lanes 16-31); N = lane&15.
  const int n  = lane & 15;
  const int mh = (lane >> 4) & 1;
#pragma unroll
  for (int j = 0; j < 4; ++j) {
    int gn = nBase + wn * 64 + j * 16 + n;
    float sag = ag[gn];
    float sau = au[gn];
#pragma unroll
    for (int i = 0; i < 2; ++i) {
      int gm0 = mBase + wm * 32 + i * 16 + mh * 8;
#pragma unroll
      for (int v = 0; v < 8; ++v) {
        float g = accG[i][j][v] * sag;
        float u = accU[i][j][v] * sau;
        float s = g / (1.0f + __expf(-g));       // silu(g)
        h[(size_t)(gm0 + v) * INTER + gn] = (__bf16)(s * u);
      }
    }
  }
}

// ------------------------- Kernel 2: down projection ------------------------
__global__ __launch_bounds__(256) void down_kernel(
    const __bf16* __restrict__ h, const float* __restrict__ Wd,
    const float* __restrict__ ad, float* __restrict__ out) {
  __shared__ __bf16 Ah[2][BM * STR];   // h tile, BM x BK (bf16, async-copied)
  __shared__ __bf16 Bd[2][BN * STR];   // Wd tile, BN x BK

  const int tid   = threadIdx.x;
  const int lane  = tid & 31;
  const int wave  = tid >> 5;
  const int wm    = wave >> 1;
  const int wn    = wave & 1;
  const int mBase = blockIdx.y * BM;
  const int nBase = blockIdx.x * BN;

  v8f acc[2][4] = {};
  float4 rw[4];

  // A tile: BM x BK bf16 = 512 x 16B chunks / 256 thr = 2 each; 4 per row.
  auto ld_async_A = [&](int buf, int k0) {
#pragma unroll
    for (int i = 0; i < 2; ++i) {
      int c = tid + i * 256;
      int row = c >> 2, c4 = c & 3;
      async_g2l_b128(Ah[buf] + row * STR + c4 * 8,
                     h + (size_t)(mBase + row) * INTER + k0 + c4 * 8);
    }
  };
  auto ld_glb_B = [&](int k0) {
#pragma unroll
    for (int i = 0; i < 4; ++i) {
      int c = tid + i * 256;
      int row = c >> 3, c4 = c & 7;
      rw[i] = *(const float4*)(Wd + (size_t)(nBase + row) * INTER + k0 + c4 * 4);
    }
  };
  auto st_lds_B = [&](int buf) {
#pragma unroll
    for (int i = 0; i < 4; ++i) {
      int c = tid + i * 256;
      int row = c >> 3, c4 = c & 7;
      *(bf4*)(Bd[buf] + row * STR + c4 * 4) = cvt_bf4(rw[i]);
    }
  };
  auto compute = [&](int buf) {
    v16bf a0 = frag_ld(Ah[buf] + (wm * 32 + 0)  * STR, lane);
    v16bf a1 = frag_ld(Ah[buf] + (wm * 32 + 16) * STR, lane);
#pragma unroll
    for (int j = 0; j < 4; ++j) {
      v16bf b = frag_ld(Bd[buf] + (wn * 64 + j * 16) * STR, lane);
      acc[0][j] = wmma_bf16(a0, b, acc[0][j]);
      acc[1][j] = wmma_bf16(a1, b, acc[1][j]);
    }
  };

  // ---- software pipeline over K (344 stages, even)
  ld_async_A(0, 0);
  ld_glb_B(0);
  st_lds_B(0);
  wait_async0();
  __syncthreads();
  const int nStages = INTER / BK;               // 344 (even)
  for (int s = 0; s < nStages; s += 2) {
    ld_async_A(1, (s + 1) * BK);                // async engine streams buf 1
    ld_glb_B((s + 1) * BK);                     //   while we WMMA buf 0
    compute(0);
    st_lds_B(1);
    wait_async0();
    __syncthreads();
    if (s + 2 < nStages) {
      ld_async_A(0, (s + 2) * BK);
      ld_glb_B((s + 2) * BK);
    }
    compute(1);
    if (s + 2 < nStages) st_lds_B(0);
    wait_async0();
    __syncthreads();
  }

  const int n  = lane & 15;
  const int mh = (lane >> 4) & 1;
#pragma unroll
  for (int j = 0; j < 4; ++j) {
    int gn = nBase + wn * 64 + j * 16 + n;
    float sad = ad[gn];
#pragma unroll
    for (int i = 0; i < 2; ++i) {
      int gm0 = mBase + wm * 32 + i * 16 + mh * 8;
#pragma unroll
      for (int v = 0; v < 8; ++v) {
        out[(size_t)(gm0 + v) * HIDDEN + gn] = acc[i][j][v] * sad;
      }
    }
  }
}

// ------------------------------- launcher ----------------------------------
extern "C" void kernel_launch(void* const* d_in, const int* in_sizes, int n_in,
                              void* d_out, int out_size, void* d_ws, size_t ws_size,
                              hipStream_t stream) {
  const float* x  = (const float*)d_in[0];
  const float* Wg = (const float*)d_in[1];
  const float* Wu = (const float*)d_in[2];
  const float* Wd = (const float*)d_in[3];
  const float* ag = (const float*)d_in[4];
  const float* au = (const float*)d_in[5];
  const float* ad = (const float*)d_in[6];
  float*  out = (float*)d_out;
  __bf16* h   = (__bf16*)d_ws;          // [8192, 11008] bf16 intermediate

  const int M = 4 * 2048;               // 8192 tokens
  dim3 block(256);
  dim3 grid1(INTER / BN, M / BM);       // 86 x 64
  gateup_swiglu_kernel<<<grid1, block, 0, stream>>>(x, Wg, Wu, ag, au, h);
  dim3 grid2(HIDDEN / BN, M / BM);      // 32 x 64
  down_kernel<<<grid2, block, 0, stream>>>(h, Wd, ad, out);
}
